// FeatureEncoder_19920058319218
// MI455X (gfx1250) — matrix operand order repaired and checked by reading, not compiled, not run on hardware
//
#include <hip/hip_runtime.h>
#include <cstdint>
#include <cstddef>

typedef __attribute__((ext_vector_type(16))) _Float16 v16h;
typedef __attribute__((ext_vector_type(8)))  _Float16 v8h;
typedef __attribute__((ext_vector_type(8)))  float    v8f;
typedef __attribute__((ext_vector_type(4)))  float    vf4;
typedef __attribute__((ext_vector_type(4)))  int      vi4;

#define GAS __attribute__((address_space(1)))
#define LAS __attribute__((address_space(3)))

#define INORM_EPS 1e-5f
#define WAVES_PER_BLOCK 8
#define BLOCK_THREADS   (WAVES_PER_BLOCK * 32)

// ---------------------------------------------------------------------------
// Weight packing: OIHW fp32 -> [CoutPad][Kpad] fp16 with k = tap*Cin + ci.
// ---------------------------------------------------------------------------
__global__ __launch_bounds__(256) void pack_weights(
    const float* __restrict__ w, _Float16* __restrict__ wpk,
    int Cin, int taps, int Cout, int Kpad, int total)
{
  int idx = blockIdx.x * 256 + threadIdx.x;
  if (idx >= total) return;
  int co = idx / Kpad;
  int k  = idx - co * Kpad;
  float v = 0.f;
  int K = taps * Cin;
  if (co < Cout && k < K) {
    int tap = k / Cin;
    int ci  = k - tap * Cin;
    v = w[((size_t)(co * Cin + ci)) * taps + tap];
  }
  wpk[idx] = (_Float16)v;
}

// ---------------------------------------------------------------------------
// Helpers
// ---------------------------------------------------------------------------
__device__ __forceinline__ int reflect_idx(int i, int n)
{
  return i < 0 ? -i : (i >= n ? 2 * n - 2 - i : i);
}

// A fragment: lane l holds row m=l&15; k-run [0..15] (lanes 0-15) and
// [16..31] (lanes 16-31) -> two contiguous 16B loads.
__device__ __forceinline__ v16h load_a_frag(const _Float16* row, int kbase, int aoff)
{
  v8h x0 = *(const v8h*)(row + kbase + 2 * aoff);
  v8h x1 = *(const v8h*)(row + kbase + 2 * aoff + 8);
  v16h a;
  #pragma unroll
  for (int q = 0; q < 8; ++q) { a[q] = x0[q]; a[8 + q] = x1[q]; }
  return a;
}

// B fragment 16-channel contiguous run: four float4 loads + cvt.
__device__ __forceinline__ void load_b_run(v16h& bb, const float* p)
{
  const vf4* p4 = (const vf4*)p;
  #pragma unroll
  for (int q = 0; q < 4; ++q) {
    vf4 f = p4[q];
    bb[4 * q + 0] = (_Float16)f[0];
    bb[4 * q + 1] = (_Float16)f[1];
    bb[4 * q + 2] = (_Float16)f[2];
    bb[4 * q + 3] = (_Float16)f[3];
  }
}

// Cooperative copy of the block's packed-weight tile into LDS using the
// CDNA5 async global->LDS DMA (ASYNCcnt) when the toolchain exposes it.
__device__ __forceinline__ void stage_weights(const _Float16* __restrict__ g,
                                              _Float16* s, int halves)
{
#if __has_builtin(__builtin_amdgcn_global_load_async_to_lds_b128)
  for (int off = threadIdx.x * 8; off < halves; off += BLOCK_THREADS * 8) {
    __builtin_amdgcn_global_load_async_to_lds_b128(
        (GAS vi4*)(g + off), (LAS vi4*)(s + off), 0, 0);
  }
#if __has_builtin(__builtin_amdgcn_s_wait_asynccnt)
  __builtin_amdgcn_s_wait_asynccnt(0);
#else
  asm volatile("s_wait_asynccnt 0x0" ::: "memory");
#endif
#else
  const v8h* gv = (const v8h*)g;
  v8h* sv = (v8h*)s;
  for (int off = threadIdx.x; off * 8 < halves; off += BLOCK_THREADS)
    sv[off] = gv[off];
#endif
  __syncthreads();
}

// ---------------------------------------------------------------------------
// Implicit-GEMM 3x3 conv (MODE 1 = stride-2 pad-1) / 2x transposed conv
// (MODE 2 = k3 s2 p1 op1), NHWC fp32 activations, f16 WMMA, fp32 accum.
// One wave: 16 pixels x (COT*16) out-channels. All geometry compile-time.
// MODE 2 tiles cover same-parity columns so parity-invalid taps (3/4 of
// them) are skipped wave-uniformly. STAGE=1: A-tile staged to LDS (async).
// ---------------------------------------------------------------------------
template<int LCIN, int MODE, int COT, int STAGE, int HIN, int HOUT, int COUT>
__global__ __launch_bounds__(BLOCK_THREADS) void conv3_wmma(
    const float* __restrict__ act, const _Float16* __restrict__ wpk,
    const float* __restrict__ bias, float* __restrict__ out)
{
  constexpr int CIN  = 1 << LCIN;
  constexpr int KPAD = (9 * CIN + 31) & ~31;
  constexpr int WIN  = HIN;
  constexpr int WOUT = HOUT;

  const int lane = threadIdx.x & 31;
  const int wave = threadIdx.x >> 5;
  const int half = lane >> 4;
  const int l15  = lane & 15;
  const int coBase = (blockIdx.y * COT) << 4;
  const int b = blockIdx.z;

  __shared__ __align__(16) _Float16 smem[STAGE ? (COT * 16 * KPAD) : 8];
  if (STAGE)
    stage_weights(wpk + (size_t)coBase * KPAD, smem, COT * 16 * KPAD);

  constexpr int tilesPerRow = WOUT >> 4;
  const int pixTile = blockIdx.x * WAVES_PER_BLOCK + wave;
  const int oy = pixTile / tilesPerRow;
  const int tr = pixTile - oy * tilesPerRow;
  int ox, px = 0;
  if (MODE == 2) {                // same-parity column tile: ox = 32g+px+2n
    px = tr & 1;
    ox = ((tr >> 1) << 5) + px + (l15 << 1);
  } else {
    ox = (tr << 4) + l15;
  }

  const _Float16* warow0 = STAGE ? (smem + (size_t)l15 * KPAD)
                                 : (wpk + (size_t)(coBase + l15) * KPAD);
  const _Float16* warow1 = STAGE ? (smem + (size_t)(16 + l15) * KPAD)
                                 : (wpk + (size_t)(coBase + 16 + l15) * KPAD);
  const int aoff = half * 8;
  const int boff = half * 16;

  v8f acc0 = {}, acc1 = {};

  if (LCIN >= 5) {
    #pragma unroll
    for (int tap = 0; tap < 9; ++tap) {
      const int ky = tap / 3;            // compile-time
      const int kx = tap - ky * 3;
      int iy;
      if (MODE == 2) {
        const int ty = oy + ky - 1;
        const int txp = px + kx - 1;
        if ((ty & 1) | (txp & 1)) continue;        // uniform parity skip
        iy = ty >> 1;
      } else {
        iy = 2 * oy + ky - 1;
      }
      if (iy < 0 || iy >= HIN) continue;           // uniform border skip
      const int ix = (MODE == 2) ? ((ox + kx - 1) >> 1) : (2 * ox + kx - 1);
      const bool ok = (ix >= 0) && (ix < WIN);
      const float* actpix = act + (((size_t)b * HIN + iy) * WIN + ix) * CIN;

      #pragma unroll
      for (int cc = 0; cc < CIN; cc += 32) {
        const int kbase = tap * CIN + cc;
        v16h bb = {};
        if (ok) load_b_run(bb, actpix + cc + boff);
        v16h av0 = load_a_frag(warow0, kbase, aoff);
        if (COT == 2) {
          acc0 = __builtin_amdgcn_wmma_f32_16x16x32_f16(false, av0, false, bb,
                                                        (short)0, acc0, false, false);
          v16h av1 = load_a_frag(warow1, kbase, aoff);
          acc1 = __builtin_amdgcn_wmma_f32_16x16x32_f16(false, av1, false, bb,
                                                        (short)0, acc1, false, false);
        } else if (((tap + (cc >> 5)) & 1) == 0) {
          acc0 = __builtin_amdgcn_wmma_f32_16x16x32_f16(false, av0, false, bb,
                                                        (short)0, acc0, false, false);
        } else {
          acc1 = __builtin_amdgcn_wmma_f32_16x16x32_f16(false, av0, false, bb,
                                                        (short)0, acc1, false, false);
        }
      }
    }
  } else {
    // CIN == 16 (enc1, stride-2 only): one 32-chunk covers taps (2t, 2t+1).
    #pragma unroll
    for (int tp = 0; tp < 5; ++tp) {
      const int kbase = tp * 32;
      const int myTap = 2 * tp + half;
      v16h bb = {};
      if (myTap < 9) {
        const int ky = myTap / 3;
        const int kx = myTap - ky * 3;
        const int iy = 2 * oy + ky - 1;
        const int ix = 2 * ox + kx - 1;
        if (iy >= 0 && iy < HIN && ix >= 0 && ix < WIN)
          load_b_run(bb, act + (((size_t)b * HIN + iy) * WIN + ix) * 16);
      }
      v16h av0 = load_a_frag(warow0, kbase, aoff);
      if (COT == 2) {
        acc0 = __builtin_amdgcn_wmma_f32_16x16x32_f16(false, av0, false, bb,
                                                      (short)0, acc0, false, false);
        v16h av1 = load_a_frag(warow1, kbase, aoff);
        acc1 = __builtin_amdgcn_wmma_f32_16x16x32_f16(false, av1, false, bb,
                                                      (short)0, acc1, false, false);
      } else if ((tp & 1) == 0) {
        acc0 = __builtin_amdgcn_wmma_f32_16x16x32_f16(false, av0, false, bb,
                                                      (short)0, acc0, false, false);
      } else {
        acc1 = __builtin_amdgcn_wmma_f32_16x16x32_f16(false, av0, false, bb,
                                                      (short)0, acc1, false, false);
      }
    }
  }

  // ---- bias + store (D rows: lane<16 -> m 0..7, lane>=16 -> m 8..15) ----
  const int mo = half * 8;
  float* op = out + ((((size_t)b * HOUT + oy) * WOUT + ox) * COUT + coBase + mo);
  if (COT == 1) acc0 += acc1;
  {
    const float* bp = bias + coBase + mo;
    vf4 bv0 = *(const vf4*)bp, bv1 = *(const vf4*)(bp + 4);
    vf4 o0, o1;
    #pragma unroll
    for (int q = 0; q < 4; ++q) { o0[q] = acc0[q] + bv0[q]; o1[q] = acc0[4 + q] + bv1[q]; }
    *(vf4*)op = o0; *(vf4*)(op + 4) = o1;
  }
  if (COT == 2) {
    const float* bp = bias + coBase + 16 + mo;
    vf4 bv0 = *(const vf4*)bp, bv1 = *(const vf4*)(bp + 4);
    vf4 o0, o1;
    #pragma unroll
    for (int q = 0; q < 4; ++q) { o0[q] = acc1[q] + bv0[q]; o1[q] = acc1[4 + q] + bv1[q]; }
    *(vf4*)(op + 16) = o0; *(vf4*)(op + 20) = o1;
  }
}

// ---------------------------------------------------------------------------
// First 7x7 conv: NCHW fp32 input (Cin=3, H=W=512), reflect pad 3, Cout=16,
// NHWC out. K=147 -> Kpad 160. Reflect tables precomputed per pixel with
// constant indexing; per-element tap/ci/ky/kx are compile-time for each
// half-wave, so gather = one select + one load per element.
// ---------------------------------------------------------------------------
__global__ __launch_bounds__(BLOCK_THREADS) void conv7_first_wmma(
    const float* __restrict__ in, const _Float16* __restrict__ wpk,
    const float* __restrict__ bias, float* __restrict__ out)
{
  constexpr int KPAD = 160, K = 147;
  const int lane = threadIdx.x & 31;
  const int wave = threadIdx.x >> 5;
  const int half = lane >> 4;
  const int l15  = lane & 15;

  const int pixTile = blockIdx.x * WAVES_PER_BLOCK + wave;
  const int oy = pixTile >> 5;                    // 512/16 = 32 tiles/row
  const int ox = ((pixTile & 31) << 4) + l15;
  const int b = blockIdx.z;
  const int bbase = b * 3 << 18;                  // b*3*512*512

  int ry[7], rx[7];
  #pragma unroll
  for (int d = 0; d < 7; ++d) {
    ry[d] = reflect_idx(oy - 3 + d, 512);
    rx[d] = reflect_idx(ox - 3 + d, 512);
  }

  const _Float16* wrow = wpk + (size_t)l15 * KPAD;
  const int aoff = half * 8;

  v8f acc0 = {}, acc1 = {};

  #pragma unroll
  for (int cidx = 0; cidx < KPAD / 32; ++cidx) {
    const int kbase = cidx * 32;
    v16h bb = {};
    #pragma unroll
    for (int j = 0; j < 16; ++j) {
      // candidate k for half 0 / half 1 -- all compile-time geometry
      const int kkA = kbase + j;            // always < K (max 143)
      const int kkB = kbase + 16 + j;
      const int tA = kkA / 3,                ciA = kkA - tA * 3;
      const int kyA = tA / 7,                kxA = tA - kyA * 7;
      const int tBc = (kkB < K) ? kkB / 3 : 0;
      const int ciB = (kkB < K) ? kkB - tBc * 3 : 0;
      const int kyB = tBc / 7,               kxB = tBc - kyB * 7;
      const int off = half ? ((ciB << 18) + (ry[kyB] << 9) + rx[kxB])
                           : ((ciA << 18) + (ry[kyA] << 9) + rx[kxA]);
      const float v = in[bbase + off];
      const bool valid = half ? (kkB < K) : true;   // compile-time per half
      bb[j] = valid ? (_Float16)v : (_Float16)0.f;
    }
    v16h av = load_a_frag(wrow, kbase, aoff);
    if ((cidx & 1) == 0)
      acc0 = __builtin_amdgcn_wmma_f32_16x16x32_f16(false, av, false, bb,
                                                    (short)0, acc0, false, false);
    else
      acc1 = __builtin_amdgcn_wmma_f32_16x16x32_f16(false, av, false, bb,
                                                    (short)0, acc1, false, false);
  }
  acc0 += acc1;

  const int mo = half * 8;
  const float* bp = bias + mo;
  float* op = out + ((((size_t)b << 9) + oy << 9) + ox << 4) + mo;
  vf4 bv0 = *(const vf4*)bp, bv1 = *(const vf4*)(bp + 4);
  vf4 o0, o1;
  #pragma unroll
  for (int q = 0; q < 4; ++q) { o0[q] = acc0[q] + bv0[q]; o1[q] = acc0[4 + q] + bv1[q]; }
  *(vf4*)op = o0; *(vf4*)(op + 4) = o1;
}

// ---------------------------------------------------------------------------
// Last 7x7 conv: NHWC fp32 in (Cin=16, H=W=512), reflect pad 3, Cout=3+tanh.
// K = 784 -> Kpad 800; each 32-chunk covers taps (2c, 2c+1) with a full
// contiguous 16-channel run -> vector loads, reflect always in-bounds.
// ---------------------------------------------------------------------------
__global__ __launch_bounds__(BLOCK_THREADS) void conv7_last_wmma(
    const float* __restrict__ in, const _Float16* __restrict__ wpk,
    const float* __restrict__ bias, float* __restrict__ out)
{
  constexpr int KPAD = 800;
  const int lane = threadIdx.x & 31;
  const int wave = threadIdx.x >> 5;
  const int half = lane >> 4;
  const int l15  = lane & 15;

  const int pixTile = blockIdx.x * WAVES_PER_BLOCK + wave;
  const int oy = pixTile >> 5;
  const int ox = ((pixTile & 31) << 4) + l15;
  const int b = blockIdx.z;

  int ry[7], rx[7];
  #pragma unroll
  for (int d = 0; d < 7; ++d) {
    ry[d] = reflect_idx(oy - 3 + d, 512);
    rx[d] = reflect_idx(ox - 3 + d, 512);
  }

  const _Float16* wrow = wpk + (size_t)l15 * KPAD;
  const int aoff = half * 8;

  v8f acc0 = {}, acc1 = {};

  #pragma unroll
  for (int cidx = 0; cidx < KPAD / 32; ++cidx) {
    const int kbase = cidx * 32;
    // 16-channel run of one tap; tap = 2*cidx (half 0) or 2*cidx+1 (half 1)
    const int tA = 2 * cidx,      kyA = tA / 7, kxA = tA - kyA * 7;
    const int tB = 2 * cidx + 1;
    const int tBc = (tB < 49) ? tB : 0;
    const int kyB = tBc / 7,      kxB = tBc - kyB * 7;
    const int iy = half ? ry[kyB] : ry[kyA];
    const int ix = half ? rx[kxB] : rx[kxA];
    v16h bb = {};
    if (!half || tB < 49)                       // compile-time per half
      load_b_run(bb, in + ((((size_t)b << 9) + iy << 9) + ix << 4));
    v16h av = load_a_frag(wrow, kbase, aoff);
    if ((cidx & 1) == 0)
      acc0 = __builtin_amdgcn_wmma_f32_16x16x32_f16(false, av, false, bb,
                                                    (short)0, acc0, false, false);
    else
      acc1 = __builtin_amdgcn_wmma_f32_16x16x32_f16(false, av, false, bb,
                                                    (short)0, acc1, false, false);
  }
  acc0 += acc1;

  const int mo = half * 8;
  const size_t pix = (((size_t)b << 9) + oy << 9) + ox;
  #pragma unroll
  for (int r = 0; r < 8; ++r) {
    const int mm = mo + r;
    if (mm < 3) out[pix * 3 + mm] = tanhf(acc0[r] + bias[mm]);
  }
}

// ---------------------------------------------------------------------------
// InstanceNorm stats / apply (NHWC, C = power of two <= 256).
// ---------------------------------------------------------------------------
__global__ __launch_bounds__(256) void inorm_stats(
    const float* __restrict__ act, float* __restrict__ stats, int HWC, int Cmask)
{
  const int t = threadIdx.x;
  const int b = blockIdx.y;
  const size_t base = (size_t)b * HWC;
  float s = 0.f, q = 0.f;
  const int e0 = blockIdx.x * 4096 + t;          // HWC multiple of 4096
  #pragma unroll
  for (int i = 0; i < 16; ++i) {
    float v = act[base + e0 + i * 256];
    s += v; q += v * v;
  }
  __shared__ float ls[256], lq[256];
  ls[t] = s; lq[t] = q;
  __syncthreads();
  for (int off = 128; off > Cmask; off >>= 1) {
    if (t < off) { ls[t] += ls[t + off]; lq[t] += lq[t + off]; }
    __syncthreads();
  }
  if (t <= Cmask) {
    atomicAdd(&stats[((size_t)b * (Cmask + 1) + t) * 2 + 0], ls[t]);
    atomicAdd(&stats[((size_t)b * (Cmask + 1) + t) * 2 + 1], lq[t]);
  }
}

__global__ __launch_bounds__(256) void inorm_apply(
    const float* __restrict__ in, const float* __restrict__ stats,
    float* __restrict__ out, size_t total, int lHWC, int Cmask, float invHW)
{
  size_t e = (size_t)blockIdx.x * 256 + threadIdx.x;
  if (e >= total) return;
  int b = (int)(e >> lHWC);
  int c = (int)e & Cmask;
  float s = stats[((size_t)b * (Cmask + 1) + c) * 2 + 0];
  float q = stats[((size_t)b * (Cmask + 1) + c) * 2 + 1];
  float mean = s * invHW;
  float var  = q * invHW - mean * mean;
  float r = rsqrtf(var + INORM_EPS);
  float v = (in[e] - mean) * r;
  out[e] = v > 0.f ? v : 0.f;
}

__global__ void zero_f32(float* p, int n)
{
  int i = blockIdx.x * 256 + threadIdx.x;
  if (i < n) p[i] = 0.f;
}

// ---------------------------------------------------------------------------
// Instance-wise segment mean over tanh output (NHWC C=3, 32 segs/batch).
// ---------------------------------------------------------------------------
__global__ __launch_bounds__(256) void seg_accum(
    const float* __restrict__ h3, const int* __restrict__ inst,
    float* __restrict__ segsum)
{
  __shared__ float ls[128];
  int t = threadIdx.x;
  if (t < 128) ls[t] = 0.f;
  __syncthreads();
  size_t p = (size_t)blockIdx.x * 256 + t;       // flat over B*H*W (HW = 2^18)
  int s = inst[p] & 31;
  atomicAdd(&ls[s * 4 + 0], h3[p * 3 + 0]);
  atomicAdd(&ls[s * 4 + 1], h3[p * 3 + 1]);
  atomicAdd(&ls[s * 4 + 2], h3[p * 3 + 2]);
  atomicAdd(&ls[s * 4 + 3], 1.f);
  __syncthreads();
  if (t < 128) {
    int b = (int)(((size_t)blockIdx.x * 256) >> 18);
    atomicAdd(&segsum[b * 128 + t], ls[t]);
  }
}

__global__ void seg_finalize(float* segsum)
{
  int t = threadIdx.x;                            // 128
  int b = blockIdx.x;                             // 4
  int c = t & 3;
  int j = t >> 2;
  if (c < 3) {
    float cnt = segsum[b * 128 + (j << 2) + 3];
    segsum[b * 128 + t] /= fmaxf(cnt, 1.f);
  }
}

__global__ __launch_bounds__(256) void seg_scatter(
    const float* __restrict__ means, const int* __restrict__ inst,
    float* __restrict__ out)
{
  size_t p = (size_t)blockIdx.x * 256 + threadIdx.x;
  int b = (int)(p >> 18);
  size_t pin = p & ((1u << 18) - 1);
  int s = inst[p] & 31;
  #pragma unroll
  for (int c = 0; c < 3; ++c)
    out[((size_t)(b * 3 + c) << 18) + pin] = means[b * 128 + s * 4 + c];
}

// ---------------------------------------------------------------------------
extern "C" void kernel_launch(void* const* d_in, const int* in_sizes, int n_in,
                              void* d_out, int out_size, void* d_ws, size_t ws_size,
                              hipStream_t stream)
{
  (void)in_sizes; (void)n_in; (void)out_size; (void)ws_size;

  const float* input = (const float*)d_in[0];
  const int*   inst  = (const int*)d_in[1];
  const float* enc_w[5]; const float* enc_b[5];
  const float* dec_w[5]; const float* dec_b[5];
  for (int i = 0; i < 5; ++i) {
    enc_w[i] = (const float*)d_in[2 + i];
    enc_b[i] = (const float*)d_in[7 + i];
    dec_w[i] = (const float*)d_in[12 + i];
    dec_b[i] = (const float*)d_in[17 + i];
  }

  const size_t ACT_BYTES = (size_t)4 * 512 * 512 * 16 * 4;   // 64 MiB
  char* ws = (char*)d_ws;
  float*    actA   = (float*)ws;
  float*    actB   = (float*)(ws + ACT_BYTES);
  _Float16* wpk    = (_Float16*)(ws + 2 * ACT_BYTES);
  float*    stats  = (float*)(ws + 2 * ACT_BYTES + (size_t)(2u << 20));
  float*    segsum = stats + 4 * 256 * 2;

  auto inorm = [&](float* act, int C, int lHWC) {
    int HWC = 1 << lHWC;
    zero_f32<<<(4 * C * 2 + 255) / 256, 256, 0, stream>>>(stats, 4 * C * 2);
    inorm_stats<<<dim3(HWC >> 12, 4), 256, 0, stream>>>(act, stats, HWC, C - 1);
    size_t total = (size_t)4 << lHWC;
    inorm_apply<<<(unsigned)(total >> 8), 256, 0, stream>>>(
        act, stats, act, total, lHWC, C - 1, 1.0f / (float)(HWC / C));
  };
  auto pack = [&](const float* w, int Cin, int taps, int Cout, int CoutPad) {
    int Kpad = (taps * Cin + 31) & ~31;
    int total = CoutPad * Kpad;
    pack_weights<<<(total + 255) / 256, 256, 0, stream>>>(w, wpk, Cin, taps, Cout, Kpad, total);
  };
  auto cgrid = [&](int Hout, int coTiles) {
    return dim3((unsigned)(((Hout * Hout) >> 4) / WAVES_PER_BLOCK), (unsigned)coTiles, 4);
  };

  // ---- encoder ----
  pack(enc_w[0], 3, 49, 16, 16);
  conv7_first_wmma<<<cgrid(512, 1), BLOCK_THREADS, 0, stream>>>(input, wpk, enc_b[0], actA);
  inorm(actA, 16, 22);

  pack(enc_w[1], 16, 9, 32, 32);
  conv3_wmma<4,1,2,1,512,256,32><<<cgrid(256, 1), BLOCK_THREADS, 0, stream>>>(actA, wpk, enc_b[1], actB);
  inorm(actB, 32, 21);

  pack(enc_w[2], 32, 9, 64, 64);
  conv3_wmma<5,1,2,1,256,128,64><<<cgrid(128, 2), BLOCK_THREADS, 0, stream>>>(actB, wpk, enc_b[2], actA);
  inorm(actA, 64, 20);

  pack(enc_w[3], 64, 9, 128, 128);
  conv3_wmma<6,1,2,1,128,64,128><<<cgrid(64, 4), BLOCK_THREADS, 0, stream>>>(actA, wpk, enc_b[3], actB);
  inorm(actB, 128, 19);

  pack(enc_w[4], 128, 9, 256, 256);
  conv3_wmma<7,1,2,0,64,32,256><<<cgrid(32, 8), BLOCK_THREADS, 0, stream>>>(actB, wpk, enc_b[4], actA);
  inorm(actA, 256, 18);

  // ---- decoder ----
  pack(dec_w[0], 256, 9, 128, 128);
  conv3_wmma<8,2,2,0,32,64,128><<<cgrid(64, 4), BLOCK_THREADS, 0, stream>>>(actA, wpk, dec_b[0], actB);
  inorm(actB, 128, 19);

  pack(dec_w[1], 128, 9, 64, 64);
  conv3_wmma<7,2,2,0,64,128,64><<<cgrid(128, 2), BLOCK_THREADS, 0, stream>>>(actB, wpk, dec_b[1], actA);
  inorm(actA, 64, 20);

  pack(dec_w[2], 64, 9, 32, 32);
  conv3_wmma<6,2,2,1,128,256,32><<<cgrid(256, 1), BLOCK_THREADS, 0, stream>>>(actA, wpk, dec_b[2], actB);
  inorm(actB, 32, 21);

  pack(dec_w[3], 32, 9, 16, 16);
  conv3_wmma<5,2,1,1,256,512,16><<<cgrid(512, 1), BLOCK_THREADS, 0, stream>>>(actB, wpk, dec_b[3], actA);
  inorm(actA, 16, 22);

  pack(dec_w[4], 16, 49, 3, 16);
  conv7_last_wmma<<<cgrid(512, 1), BLOCK_THREADS, 0, stream>>>(actA, wpk, dec_b[4], actB);

  // ---- instance-wise segment mean -> NCHW output ----
  zero_f32<<<2, 256, 0, stream>>>(segsum, 512);
  seg_accum<<<4096, 256, 0, stream>>>(actB, inst, segsum);
  seg_finalize<<<4, 128, 0, stream>>>(segsum);
  seg_scatter<<<4096, 256, 0, stream>>>(segsum, inst, (float*)d_out);
}
